// LoraLayer_206158430664
// MI455X (gfx1250) — compile-verified, matrix operand order checked by reading, not run on hardware
//
#include <hip/hip_runtime.h>

// ---------------------------------------------------------------------------
// Grouped LoRA (SGMV-style) for MI455X / gfx1250, wave32 + WMMA f16.
// out[t, m*O+o] = (x[t] . A[m,a_t]) . B[m,a_t], rank-masked per adapter.
// Memory-bound problem (out alone is 384MB); compute in f16 WMMA, f32 accum.
// ---------------------------------------------------------------------------

#define T_DIM 8192
#define H_DIM 4096
#define M_DIM 3
#define A_DIM 8
#define R_DIM 64
#define O_DIM 4096

typedef __attribute__((ext_vector_type(16))) _Float16 v16h;
typedef __attribute__((ext_vector_type(8)))  _Float16 v8h;
typedef __attribute__((ext_vector_type(4)))  _Float16 v4h;
typedef __attribute__((ext_vector_type(8)))  float    v8f;

// Build a 16x32 f16 WMMA fragment for this lane (ISA 7.12.2 layout):
// lane L holds row (L&15); VGPR0..3 = K[0..7] (+8 if L>=16), VGPR4..7 = K[16..23] (+8).
// Caller passes p already offset by (lane>>4)*8 halves; rows are K-contiguous.
__device__ __forceinline__ v16h load_frag(const _Float16* __restrict__ p) {
    v8h lo = *reinterpret_cast<const v8h*>(p);        // K + 0..7
    v8h hi = *reinterpret_cast<const v8h*>(p + 16);   // K + 16..23
    v16h r;
#pragma unroll
    for (int i = 0; i < 8; ++i) { r[i] = lo[i]; r[i + 8] = hi[i]; }
    return r;
}

// ---- prep: x f32 -> f16 ----------------------------------------------------
__global__ void k_cvt_x(const float* __restrict__ x, _Float16* __restrict__ xh) {
    int i = (blockIdx.x * blockDim.x + threadIdx.x) * 4;   // T*H divisible by 4
    float4 v = *reinterpret_cast<const float4*>(x + i);
    v4h o = { (_Float16)v.x, (_Float16)v.y, (_Float16)v.z, (_Float16)v.w };
    *reinterpret_cast<v4h*>(xh + i) = o;
}

// ---- prep: lora_a [m,a,h,r] f32 -> Ah [m,a,r,h] f16, rank-masked -----------
__global__ void k_cvt_a(const float* __restrict__ la, _Float16* __restrict__ Ah,
                        const int* __restrict__ ranks) {
    int tid = blockIdx.x * blockDim.x + threadIdx.x;   // over M*A*H*R
    int r  = tid & (R_DIM - 1);
    int h  = (tid >> 6) & (H_DIM - 1);                 // R_DIM == 64
    int ma = tid >> 18;                                // H*R == 2^18
    int a  = ma & (A_DIM - 1);
    _Float16 hv = (r < ranks[a]) ? (_Float16)la[tid] : (_Float16)0.0f;
    Ah[((size_t)ma * R_DIM + r) * H_DIM + h] = hv;     // zero rows r>=rank => free masking
}

// ---- prep: lora_b [m,a,r,o] f32 -> Bh [m,a,o,r] f16 ------------------------
__global__ void k_cvt_b(const float* __restrict__ lb, _Float16* __restrict__ Bh) {
    int tid = blockIdx.x * blockDim.x + threadIdx.x;   // over M*A*R*O
    int o  = tid & (O_DIM - 1);
    int r  = (tid >> 12) & (R_DIM - 1);                // O_DIM == 4096
    int ma = tid >> 18;                                // R*O == 2^18
    Bh[((size_t)ma * O_DIM + o) * R_DIM + r] = (_Float16)lb[tid];
}

// ---- bucket tokens by adapter ----------------------------------------------
__global__ void k_group(const int* __restrict__ ids, int* __restrict__ grp,
                        int* __restrict__ cnt) {
    int t = blockIdx.x * blockDim.x + threadIdx.x;
    if (t < T_DIM) {
        int a = ids[t];
        int pos = atomicAdd(&cnt[a], 1);
        grp[a * T_DIM + pos] = t;
    }
}

// ---- phase 1: shrink. inter[m,t,r] = x[t,:] @ A[m,a,:,r] -------------------
// block = 128 thr = 4 waves; wave w owns r in [16w,16w+16); 16 gathered tokens.
__global__ void k_shrink(const _Float16* __restrict__ xh,
                         const _Float16* __restrict__ Ah,
                         _Float16* __restrict__ inter,
                         const int* __restrict__ grp,
                         const int* __restrict__ cnt) {
    const int a     = blockIdx.y;
    const int tbase = blockIdx.x * 16;
    const int c     = cnt[a];
    if (tbase >= c) return;

    const int wave = threadIdx.x >> 5;
    const int lane = threadIdx.x & 31;
    const int hs   = lane >> 4;          // K-half select
    const int lrow = lane & 15;          // fragment row
    const int gbase = a * T_DIM + tbase;

    const int tokA = (tbase + lrow < c) ? grp[gbase + lrow] : grp[gbase];
    const _Float16* xrow = xh + (size_t)tokA * H_DIM + hs * 8;

    const int r0 = wave * 16;
    const _Float16* w0 = Ah + ((size_t)(0 * A_DIM + a) * R_DIM + r0 + lrow) * H_DIM + hs * 8;
    const _Float16* w1 = Ah + ((size_t)(1 * A_DIM + a) * R_DIM + r0 + lrow) * H_DIM + hs * 8;
    const _Float16* w2 = Ah + ((size_t)(2 * A_DIM + a) * R_DIM + r0 + lrow) * H_DIM + hs * 8;

    v8f acc0 = {}, acc1 = {}, acc2 = {};
#pragma unroll 2
    for (int k = 0; k < H_DIM; k += 32) {
        v16h af = load_frag(xrow + k);   // A: 16 tokens x 32 h
        v16h b0 = load_frag(w0 + k);     // B: 32 h x 16 r (rows pre-transposed)
        v16h b1 = load_frag(w1 + k);
        v16h b2 = load_frag(w2 + k);
        acc0 = __builtin_amdgcn_wmma_f32_16x16x32_f16(false, af, false, b0, (short)0, acc0, false, false);
        acc1 = __builtin_amdgcn_wmma_f32_16x16x32_f16(false, af, false, b1, (short)0, acc1, false, false);
        acc2 = __builtin_amdgcn_wmma_f32_16x16x32_f16(false, af, false, b2, (short)0, acc2, false, false);
    }

    // D layout: VGPR v, lanes[0:16) -> (row v, col lane); lanes[16:32) -> (row v+8).
    int  toks[8]; bool val[8];
#pragma unroll
    for (int v = 0; v < 8; ++v) {
        int row  = v + 8 * hs;
        int tIdx = tbase + row;
        val[v]  = tIdx < c;
        toks[v] = val[v] ? grp[gbase + row] : 0;
    }
#pragma unroll
    for (int v = 0; v < 8; ++v) {
        if (val[v]) {
            size_t base = (size_t)toks[v] * R_DIM + r0 + lrow;
            inter[(size_t)0 * T_DIM * R_DIM + base] = (_Float16)acc0[v];
            inter[(size_t)1 * T_DIM * R_DIM + base] = (_Float16)acc1[v];
            inter[(size_t)2 * T_DIM * R_DIM + base] = (_Float16)acc2[v];
        }
    }
}

// ---- phase 2: expand. out[t, m*O+o] = inter[m,t,:] @ B[m,a,:,o] ------------
// block = 256 thr = 8 waves sweeping the 256 o-tiles per module; K = R = 64.
__global__ void k_expand(const _Float16* __restrict__ inter,
                         const _Float16* __restrict__ Bh,
                         float* __restrict__ out,
                         const int* __restrict__ grp,
                         const int* __restrict__ cnt) {
    const int a     = blockIdx.y;
    const int tbase = blockIdx.x * 16;
    const int c     = cnt[a];
    if (tbase >= c) return;

    const int wave = threadIdx.x >> 5;
    const int lane = threadIdx.x & 31;
    const int hs   = lane >> 4;
    const int lrow = lane & 15;
    const int gbase = a * T_DIM + tbase;

    const int tokA = (tbase + lrow < c) ? grp[gbase + lrow] : grp[gbase];

    int  toks[8]; bool val[8];
#pragma unroll
    for (int v = 0; v < 8; ++v) {
        int row  = v + 8 * hs;
        int tIdx = tbase + row;
        val[v]  = tIdx < c;
        toks[v] = val[v] ? grp[gbase + row] : 0;
    }

#pragma unroll
    for (int m = 0; m < M_DIM; ++m) {
        const _Float16* ib = inter + ((size_t)m * T_DIM + tokA) * R_DIM + hs * 8;
        v16h a0 = load_frag(ib);          // K = 0..31
        v16h a1 = load_frag(ib + 32);     // K = 32..63

        for (int ot = wave; ot < O_DIM / 16; ot += 8) {
            const int o0 = ot * 16;
            const _Float16* wrow =
                Bh + ((size_t)(m * A_DIM + a) * O_DIM + o0 + lrow) * R_DIM + hs * 8;
            __builtin_prefetch(wrow + 8 * 16 * R_DIM, 0, 1);  // next tile for this wave
            v16h b0 = load_frag(wrow);
            v16h b1 = load_frag(wrow + 32);
            v8f acc = {};
            acc = __builtin_amdgcn_wmma_f32_16x16x32_f16(false, a0, false, b0, (short)0, acc, false, false);
            acc = __builtin_amdgcn_wmma_f32_16x16x32_f16(false, a1, false, b1, (short)0, acc, false, false);
#pragma unroll
            for (int v = 0; v < 8; ++v) {
                if (val[v]) {
                    out[(size_t)toks[v] * (M_DIM * O_DIM) + (size_t)m * O_DIM + o0 + lrow]
                        = acc[v];
                }
            }
        }
    }
}

// ---------------------------------------------------------------------------
extern "C" void kernel_launch(void* const* d_in, const int* in_sizes, int n_in,
                              void* d_out, int out_size, void* d_ws, size_t ws_size,
                              hipStream_t stream) {
    (void)in_sizes; (void)n_in; (void)out_size;
    const float* x     = (const float*)d_in[0];
    const float* la    = (const float*)d_in[1];
    const float* lb    = (const float*)d_in[2];
    const int*   ids   = (const int*)d_in[3];
    const int*   ranks = (const int*)d_in[4];
    float* out = (float*)d_out;

    // Workspace layout (all offsets 128B-aligned):
    const size_t xh_bytes    = (size_t)T_DIM * H_DIM * 2;                 // 64 MiB
    const size_t Ah_bytes    = (size_t)M_DIM * A_DIM * R_DIM * H_DIM * 2; // 12 MiB
    const size_t Bh_bytes    = (size_t)M_DIM * A_DIM * O_DIM * R_DIM * 2; // 12 MiB
    const size_t inter_bytes = (size_t)M_DIM * T_DIM * R_DIM * 2;         // 3 MiB
    const size_t grp_bytes   = (size_t)A_DIM * T_DIM * sizeof(int);       // 256 KiB
    const size_t need = xh_bytes + Ah_bytes + Bh_bytes + inter_bytes + grp_bytes
                      + A_DIM * sizeof(int);
    if (ws_size < need) return;  // cannot run safely without scratch

    char* ws = (char*)d_ws;
    _Float16* xh    = (_Float16*)(ws);
    _Float16* Ah    = (_Float16*)(ws + xh_bytes);
    _Float16* Bh    = (_Float16*)(ws + xh_bytes + Ah_bytes);
    _Float16* inter = (_Float16*)(ws + xh_bytes + Ah_bytes + Bh_bytes);
    int* grp = (int*)(ws + xh_bytes + Ah_bytes + Bh_bytes + inter_bytes);
    int* cnt = grp + (size_t)A_DIM * T_DIM;

    hipMemsetAsync(cnt, 0, A_DIM * sizeof(int), stream);

    k_cvt_x<<<(T_DIM * H_DIM) / 4 / 256, 256, 0, stream>>>(x, xh);
    k_cvt_a<<<(M_DIM * A_DIM * H_DIM * R_DIM) / 256, 256, 0, stream>>>(la, Ah, ranks);
    k_cvt_b<<<(M_DIM * A_DIM * R_DIM * O_DIM) / 256, 256, 0, stream>>>(lb, Bh);
    k_group<<<T_DIM / 256, 256, 0, stream>>>(ids, grp, cnt);

    dim3 tiles(T_DIM / 16, A_DIM);
    k_shrink<<<tiles, 128, 0, stream>>>(xh, Ah, inter, grp, cnt);
    k_expand<<<tiles, 256, 0, stream>>>(inter, Bh, out, grp, cnt);
}
